// MultiHeadAttention_47132971106400
// MI455X (gfx1250) — compile-verified, matrix operand order checked by reading, not run on hardware
//
#include <hip/hip_runtime.h>
#include <hip/hip_bf16.h>

// ---------------------------------------------------------------------------
// MHA for B=2, S=2048, D=1024, H=16, HD=64 on gfx1250 (MI455X).
// f16 WMMA (v_wmma_f32_16x16x32_f16), fp32 accumulate.
// Projections: BK=64, double-buffered LDS, 1 barrier per K-step.
// Attention: flash-style, async global->LDS K/V staging (ASYNCcnt),
// double-buffered, bit-packed mask, in-place f16 score/prob buffer.
// ---------------------------------------------------------------------------

#define BATCH 2
#define SEQ   2048
#define DMODEL 1024
#define NHEAD 16
#define HDIM  64
#define MROWS (BATCH * SEQ)   // 4096
#define SWORDS (SEQ / 32)     // packed mask words per row = 64

typedef __attribute__((ext_vector_type(16))) _Float16 v16h;
typedef __attribute__((ext_vector_type(8)))  _Float16 v8h;
typedef __attribute__((ext_vector_type(8)))  float    v8f;

__device__ __forceinline__ v8f wmma_f16(v16h a, v16h b, v8f c) {
  return __builtin_amdgcn_wmma_f32_16x16x32_f16(
      false, a, false, b, (short)0, c, false, false);
}

// 16x32 f16 fragment from LDS tile [row][k], `stride` halves/row (mult of 8).
__device__ __forceinline__ v16h load_frag(const _Float16* base, int row0,
                                          int k0, int stride) {
  const int lane = threadIdx.x & 31;
  const _Float16* p =
      base + (row0 + (lane & 15)) * stride + k0 + ((lane >> 4) << 3);
  v8h lo = *(const v8h*)p;
  v8h hi = *(const v8h*)(p + 16);
  v16h r;
#pragma unroll
  for (int i = 0; i < 8; ++i) { r[i] = lo[i]; r[i + 8] = hi[i]; }
  return r;
}

// Async global -> LDS copy, 16B per lane.  Tracked with ASYNCcnt.
// LDS byte offset = low 32 bits of the generic address of a __shared__ ptr.
__device__ __forceinline__ void async_cp16(_Float16* lds, const _Float16* g) {
  unsigned loff = (unsigned)(size_t)lds;
  unsigned long long ga = (unsigned long long)(size_t)g;
  asm volatile("global_load_async_to_lds_b128 %0, %1, off"
               :: "v"(loff), "v"(ga) : "memory");
}

// ---- register staging helpers for the projection GEMM ----------------------
__device__ __forceinline__ void ldA32(const float* Af, int row0, int k0,
                                      int tid, float4 (&ar)[8]) {
#pragma unroll
  for (int i = 0; i < 8; ++i) {
    int e = tid + (i << 8), r = e >> 4, c = (e << 2) & 63;
    ar[i] = *(const float4*)(Af + (size_t)(row0 + r) * DMODEL + k0 + c);
  }
}
__device__ __forceinline__ void stA32(_Float16* As, int tid,
                                      const float4 (&ar)[8]) {
#pragma unroll
  for (int i = 0; i < 8; ++i) {
    int e = tid + (i << 8), r = e >> 4, c = (e << 2) & 63;
    _Float16* d = &As[r * 72 + c];
    d[0] = (_Float16)ar[i].x; d[1] = (_Float16)ar[i].y;
    d[2] = (_Float16)ar[i].z; d[3] = (_Float16)ar[i].w;
  }
}
__device__ __forceinline__ void ldA16(const _Float16* Ah, int row0, int k0,
                                      int tid, v8h (&ar)[4]) {
#pragma unroll
  for (int i = 0; i < 4; ++i) {
    int e = tid + (i << 8), r = e >> 3, c = (e << 3) & 63;
    ar[i] = *(const v8h*)(Ah + (size_t)(row0 + r) * DMODEL + k0 + c);
  }
}
__device__ __forceinline__ void stA16(_Float16* As, int tid,
                                      const v8h (&ar)[4]) {
#pragma unroll
  for (int i = 0; i < 4; ++i) {
    int e = tid + (i << 8), r = e >> 3, c = (e << 3) & 63;
    *(v8h*)&As[r * 72 + c] = ar[i];
  }
}
__device__ __forceinline__ void ldB(const float* W, int col0, int k0, int tid,
                                    float4 (&br)[4]) {
#pragma unroll
  for (int i = 0; i < 4; ++i) {
    int e = tid + (i << 8), kr = e >> 4, nc = (e << 2) & 63;
    br[i] = *(const float4*)(W + (size_t)(k0 + kr) * DMODEL + col0 + nc);
  }
}
__device__ __forceinline__ void stB(_Float16* Bs, int tid,
                                    const float4 (&br)[4]) {
#pragma unroll
  for (int i = 0; i < 4; ++i) {
    int e = tid + (i << 8), kr = e >> 4, nc = (e << 2) & 63;
    Bs[(nc + 0) * 72 + kr] = (_Float16)br[i].x;   // transposed [n][k]
    Bs[(nc + 1) * 72 + kr] = (_Float16)br[i].y;
    Bs[(nc + 2) * 72 + kr] = (_Float16)br[i].z;
    Bs[(nc + 3) * 72 + kr] = (_Float16)br[i].w;
  }
}

// ---------------------------------------------------------------------------
// C[4096x1024] = (A @ W + bias) * scale.  Tile 128x64, BK=64, double-buffered.
// MODE 0: f16 out [b,h,s,d]; MODE 1: f16 out [b,h,d,s]; MODE 2: f32 out.
// ---------------------------------------------------------------------------
template <int MODE, bool AHALF>
__global__ __launch_bounds__(256) void proj_kernel(
    const void* __restrict__ Av, const float* __restrict__ W,
    const float* __restrict__ bias, void* __restrict__ outv, float scale) {
  __shared__ _Float16 As[2][128 * 72];
  __shared__ _Float16 Bs[2][64 * 72];

  const int tid = threadIdx.x;
  const int wave = tid >> 5, lane = tid & 31;
  const int wm = wave & 3, wn = wave >> 2;
  const int row0 = blockIdx.y * 128;
  const int col0 = blockIdx.x * 64;

  const float*    Af = (const float*)Av;
  const _Float16* Ah = (const _Float16*)Av;

  float4 ar[8]; v8h arh[4]; float4 br[4];
  v8f acc[2][2] = {};

  if (AHALF) ldA16(Ah, row0, 0, tid, arh); else ldA32(Af, row0, 0, tid, ar);
  ldB(W, col0, 0, tid, br);
  if (AHALF) stA16(As[0], tid, arh); else stA32(As[0], tid, ar);
  stB(Bs[0], tid, br);
  __syncthreads();

  for (int kt = 0; kt < DMODEL / 64; ++kt) {
    const int cur = kt & 1;
    const bool nxt = (kt + 1) < DMODEL / 64;
    if (nxt) {  // overlap next tile's global loads with this tile's WMMAs
      int k0n = (kt + 1) * 64;
      if (AHALF) ldA16(Ah, row0, k0n, tid, arh);
      else       ldA32(Af, row0, k0n, tid, ar);
      ldB(W, col0, k0n, tid, br);
    }
#pragma unroll
    for (int kk = 0; kk < 64; kk += 32) {
      v16h a0 = load_frag(As[cur], wm * 32 + 0,  kk, 72);
      v16h a1 = load_frag(As[cur], wm * 32 + 16, kk, 72);
      v16h b0 = load_frag(Bs[cur], wn * 32 + 0,  kk, 72);
      v16h b1 = load_frag(Bs[cur], wn * 32 + 16, kk, 72);
      acc[0][0] = wmma_f16(a0, b0, acc[0][0]);
      acc[0][1] = wmma_f16(a0, b1, acc[0][1]);
      acc[1][0] = wmma_f16(a1, b0, acc[1][0]);
      acc[1][1] = wmma_f16(a1, b1, acc[1][1]);
    }
    if (nxt) {
      if (AHALF) stA16(As[cur ^ 1], tid, arh); else stA32(As[cur ^ 1], tid, ar);
      stB(Bs[cur ^ 1], tid, br);
      __syncthreads();
    }
  }

  const int lhi = (lane >> 4) << 3, llo = lane & 15;
#pragma unroll
  for (int mi = 0; mi < 2; ++mi)
#pragma unroll
    for (int ni = 0; ni < 2; ++ni)
#pragma unroll
      for (int v = 0; v < 8; ++v) {
        int r = row0 + wm * 32 + mi * 16 + v + lhi;
        int c = col0 + wn * 32 + ni * 16 + llo;
        float val = (acc[mi][ni][v] + bias[c]) * scale;
        if (MODE == 2) {
          ((float*)outv)[(size_t)r * DMODEL + c] = val;
        } else {
          int b = r >> 11, s = r & 2047;
          int h = c >> 6,  d = c & 63;
          size_t off =
              (MODE == 0)
                  ? (((size_t)(b * NHEAD + h) * SEQ + s) * HDIM + d)
                  : (((size_t)(b * NHEAD + h) * HDIM + d) * SEQ + s);
          ((_Float16*)outv)[off] = (_Float16)val;
        }
      }
}

// ---------------------------------------------------------------------------
// Pack int32 mask into bits: one wave32 ballot per 32 ints.
// ---------------------------------------------------------------------------
__global__ __launch_bounds__(256) void mask_pack_kernel(
    const int* __restrict__ mask, unsigned* __restrict__ pm) {
  size_t i = (size_t)blockIdx.x * 256 + threadIdx.x;
  unsigned long long bal = __ballot(mask[i] != 0);
  if ((threadIdx.x & 31) == 0) pm[i >> 5] = (unsigned)bal;
}

// ---------------------------------------------------------------------------
// Flash attention.  Block = (b,h, 64-query tile); 32 key tiles of 64.
// K/V staged via async global->LDS copies, double-buffered (ASYNCcnt pipeline).
// Scores/probs share one in-place f16 LDS buffer.
// ---------------------------------------------------------------------------
__global__ __launch_bounds__(256) void attn_kernel(
    const _Float16* __restrict__ qh,    // [b,h,s,d]  (pre-scaled by 1/8)
    const _Float16* __restrict__ kh,    // [b,h,s,d]
    const _Float16* __restrict__ vt,    // [b,h,d,s]
    const unsigned* __restrict__ pmask, // [S, S/32] bit-packed
    _Float16* __restrict__ ctx) {       // [b,s,h*64]
  __shared__ _Float16 Qs[64 * 72];
  __shared__ _Float16 Ks[2][64 * 72];  // [key][d], double-buffered
  __shared__ _Float16 Vs[2][64 * 72];  // [d][key], double-buffered
  __shared__ _Float16 SP[64 * 72];     // scores then probs, in place (f16)
  __shared__ float rowm[64], rowl[64], alpha[64];
  __shared__ float pmax[64 * 4], psum[64 * 4];

  const int tid = threadIdx.x, lane = tid & 31, wave = tid >> 5;
  const int wm = wave & 3, wn = wave >> 2;
  const int lhi = (lane >> 4) << 3, llo = lane & 15;
  const int bh = blockIdx.y, b = bh >> 4, h = bh & 15;
  const int q0 = blockIdx.x * 64;
  const int sr = tid >> 2, sq = tid & 3, scb = sq << 4;

  const _Float16* qbase = qh + (size_t)bh * SEQ * HDIM;
  const _Float16* kbase = kh + (size_t)bh * SEQ * HDIM;
  const _Float16* vbase = vt + (size_t)bh * HDIM * SEQ;

  const int ldr = tid >> 3, ldc = (tid & 7) << 3;  // staging row / col

  for (int e = tid; e < 512; e += 256) {
    int r = e >> 3, c = (e & 7) << 3;
    *(v8h*)&Qs[r * 72 + c] = *(const v8h*)(qbase + (size_t)(q0 + r) * HDIM + c);
  }
  if (tid < 64) { rowm[tid] = -3.0e38f; rowl[tid] = 0.f; }
  v8f o[2] = {};

  // prologue: async-stage key tile 0 into buffer 0 (4 ops/wave outstanding)
  async_cp16(&Ks[0][ldr * 72 + ldc], kbase + (size_t)ldr * HDIM + ldc);
  async_cp16(&Ks[0][(ldr + 32) * 72 + ldc],
             kbase + (size_t)(ldr + 32) * HDIM + ldc);
  async_cp16(&Vs[0][ldr * 72 + ldc], vbase + (size_t)ldr * SEQ + ldc);
  async_cp16(&Vs[0][(ldr + 32) * 72 + ldc],
             vbase + (size_t)(ldr + 32) * SEQ + ldc);

  for (int kt = 0; kt < SEQ / 64; ++kt) {
    const int cur = kt & 1;
    if (kt + 1 < SEQ / 64) {  // issue next tile into the other buffer
      const int k1 = (kt + 1) * 64;
      async_cp16(&Ks[cur ^ 1][ldr * 72 + ldc],
                 kbase + (size_t)(k1 + ldr) * HDIM + ldc);
      async_cp16(&Ks[cur ^ 1][(ldr + 32) * 72 + ldc],
                 kbase + (size_t)(k1 + ldr + 32) * HDIM + ldc);
      async_cp16(&Vs[cur ^ 1][ldr * 72 + ldc],
                 vbase + (size_t)ldr * SEQ + k1 + ldc);
      async_cp16(&Vs[cur ^ 1][(ldr + 32) * 72 + ldc],
                 vbase + (size_t)(ldr + 32) * SEQ + k1 + ldc);
      asm volatile("s_wait_asynccnt 0x4" ::: "memory");  // tile kt landed
    } else {
      asm volatile("s_wait_asynccnt 0x0" ::: "memory");
    }
    __syncthreads();

    // ---- S = Q @ K^T (Q pre-scaled by 1/sqrt(HD)) ----
    const int k0 = kt * 64;
    v8f s[2] = {};
#pragma unroll
    for (int kk = 0; kk < 64; kk += 32) {
      v16h a  = load_frag(Qs,      wm * 16,      kk, 72);
      v16h b0 = load_frag(Ks[cur], wn * 32 + 0,  kk, 72);
      v16h b1 = load_frag(Ks[cur], wn * 32 + 16, kk, 72);
      s[0] = wmma_f16(a, b0, s[0]);
      s[1] = wmma_f16(a, b1, s[1]);
    }
    // ---- mask via packed bits -> f16 scores in SP ----
#pragma unroll
    for (int v = 0; v < 8; ++v) {
      int r = wm * 16 + v + lhi;
      unsigned wbits = pmask[(size_t)(q0 + r) * SWORDS + (k0 >> 5) + wn];
      SP[r * 72 + wn * 32 + llo] =
          (_Float16)(((wbits >> llo) & 1u) ? s[0][v] : -6.0e4f);
      SP[r * 72 + wn * 32 + 16 + llo] =
          (_Float16)(((wbits >> (llo + 16)) & 1u) ? s[1][v] : -6.0e4f);
    }
    __syncthreads();

    // ---- online softmax, 4 threads per row, exp in place (f16) ----
    float mseg = -3.0e38f;
#pragma unroll
    for (int c = 0; c < 16; ++c)
      mseg = fmaxf(mseg, (float)SP[sr * 72 + scb + c]);
    pmax[(sr << 2) + sq] = mseg;
    __syncthreads();
    float mold = rowm[sr];
    float mnew = fmaxf(mold,
        fmaxf(fmaxf(pmax[sr << 2], pmax[(sr << 2) + 1]),
              fmaxf(pmax[(sr << 2) + 2], pmax[(sr << 2) + 3])));
    float sum = 0.f;
#pragma unroll
    for (int c = 0; c < 16; ++c) {
      float p = __expf((float)SP[sr * 72 + scb + c] - mnew);
      SP[sr * 72 + scb + c] = (_Float16)p;   // own cells only
      sum += p;
    }
    psum[(sr << 2) + sq] = sum;
    float al = __expf(mold - mnew);
    alpha[sr] = al;          // all 4 row-threads write identical value
    __syncthreads();
    if (sq == 0) {
      rowl[sr] = rowl[sr] * al + psum[sr << 2] + psum[(sr << 2) + 1] +
                 psum[(sr << 2) + 2] + psum[(sr << 2) + 3];
      rowm[sr] = mnew;
    }

    // ---- rescale accumulator, O += P @ V ----
#pragma unroll
    for (int v = 0; v < 8; ++v) {
      float a2 = alpha[wm * 16 + v + lhi];
      o[0][v] *= a2;
      o[1][v] *= a2;
    }
#pragma unroll
    for (int kk = 0; kk < 64; kk += 32) {
      v16h a  = load_frag(SP,      wm * 16,      kk, 72);
      v16h b0 = load_frag(Vs[cur], wn * 32 + 0,  kk, 72);
      v16h b1 = load_frag(Vs[cur], wn * 32 + 16, kk, 72);
      o[0] = wmma_f16(a, b0, o[0]);
      o[1] = wmma_f16(a, b1, o[1]);
    }
    __syncthreads();
  }

  if (tid < 64) alpha[tid] = 1.0f / rowl[tid];
  __syncthreads();

#pragma unroll
  for (int ni = 0; ni < 2; ++ni)
#pragma unroll
    for (int v = 0; v < 8; ++v) {
      int r = wm * 16 + v + lhi;
      int c = wn * 32 + ni * 16 + llo;
      float val = o[ni][v] * alpha[r];
      size_t off = ((size_t)(b * SEQ + q0 + r)) * DMODEL + h * HDIM + c;
      ctx[off] = (_Float16)val;
    }
}

// ---------------------------------------------------------------------------
extern "C" void kernel_launch(void* const* d_in, const int* in_sizes, int n_in,
                              void* d_out, int out_size, void* d_ws,
                              size_t ws_size, hipStream_t stream) {
  const float* Q   = (const float*)d_in[0];
  const float* K   = (const float*)d_in[1];
  const float* V   = (const float*)d_in[2];
  const int*   msk = (const int*)  d_in[3];
  const float* Wq  = (const float*)d_in[4];
  const float* bq  = (const float*)d_in[5];
  const float* Wk  = (const float*)d_in[6];
  const float* bk  = (const float*)d_in[7];
  const float* Wv  = (const float*)d_in[8];
  const float* bv  = (const float*)d_in[9];
  const float* Wo  = (const float*)d_in[10];
  const float* bo  = (const float*)d_in[11];
  float* out = (float*)d_out;

  // Workspace: 4x 8MB f16 tensors + 512KB packed mask
  char* ws = (char*)d_ws;
  _Float16* qh  = (_Float16*)(ws + (size_t)0  * (1 << 20));
  _Float16* khd = (_Float16*)(ws + (size_t)8  * (1 << 20));
  _Float16* vtd = (_Float16*)(ws + (size_t)16 * (1 << 20));
  _Float16* ctx = (_Float16*)(ws + (size_t)24 * (1 << 20));
  unsigned* pm  = (unsigned*)(ws + (size_t)32 * (1 << 20));

  mask_pack_kernel<<<(SEQ * SEQ) / 256, 256, 0, stream>>>(msk, pm);

  dim3 pg(DMODEL / 64, MROWS / 128);  // (16, 32)
  proj_kernel<0, false><<<pg, 256, 0, stream>>>(Q, Wq, bq, qh, 0.125f);
  proj_kernel<0, false><<<pg, 256, 0, stream>>>(K, Wk, bk, khd, 1.0f);
  proj_kernel<1, false><<<pg, 256, 0, stream>>>(V, Wv, bv, vtd, 1.0f);

  dim3 ag(SEQ / 64, BATCH * NHEAD);   // (32, 32)
  attn_kernel<<<ag, 256, 0, stream>>>(qh, khd, vtd, pm, ctx);

  proj_kernel<2, true><<<pg, 256, 0, stream>>>(ctx, Wo, bo, out, 1.0f);
}
